// MLP_Poisson_23390391894489
// MI455X (gfx1250) — compile-verified, matrix operand order checked by reading, not run on hardware
//
#include <hip/hip_runtime.h>
#include <hip/hip_bf16.h>
#include <stdint.h>

typedef __attribute__((ext_vector_type(16))) _Float16 v16h;
typedef __attribute__((ext_vector_type(8)))  float    v8f;

#define LEAK     0.95f
#define KDIM     784        // input features
#define NHID     800        // hidden units
#define KT_TILES 25         // K padded 784 -> 800 = 25 tiles of 32
#define NT_TILES 50         // 800 / 16
#define BATCH    8192
#define BT       16         // batch rows per workgroup (one WMMA M-tile)
#define WAVES    8
#define THREADS  (WAVES * 32)

#define WMMA_F16(a, b, c) \
  __builtin_amdgcn_wmma_f32_16x16x32_f16(false, (a), false, (b), (short)0, (c), false, false)

// CDNA5 16-bit A/B fragment K index for (lane, half) per ISA 7.12.2:
// VGPR0: K=0,1 (lanes 0-15) / K=8,9 (lanes 16-31); VGPR1: K=2,3 / 10,11; ...
// VGPR4-7: +16.
__device__ __forceinline__ int kmap(int lane, int h) {
  int hi = (lane >> 4) & 1;
  int vg = h >> 1;
  int lo = h & 1;
  return ((vg >= 4) ? 16 : 0) + ((vg & 3) << 1) + lo + (hi << 3);
}

// Stateless hash RNG -> uniform [0,1)
__device__ __forceinline__ float rng01(uint32_t row, uint32_t k, uint32_t step) {
  uint32_t x = row * 784u + k;
  x += (step + 1u) * 0x9E3779B9u;
  x ^= x >> 16; x *= 0x7FEB352Du;
  x ^= x >> 15; x *= 0x846CA68Bu;
  x ^= x >> 16;
  return (float)(x >> 8) * (1.0f / 16777216.0f);
}

// Pack w1 [800,784] f32 -> f16 B-fragments, layout [nt][kt][lane][16 halves]
// (per-N-tile contiguous 25.6KB streams -> immediate-offset loads in main loop).
// K zero-padded 784 -> 800.
__global__ void pack_w1_kernel(const float* __restrict__ w1,
                               _Float16* __restrict__ w1h) {
  int idx = blockIdx.x * blockDim.x + threadIdx.x;
  if (idx >= NT_TILES * KT_TILES * 32 * 16) return;
  int h  = idx & 15;
  int t  = idx >> 4;
  int ln = t & 31; t >>= 5;
  int kt = t % KT_TILES;
  int nt = t / KT_TILES;
  int n = nt * 16 + (ln & 15);
  int k = kt * 32 + kmap(ln, h);
  float v = (k < KDIM) ? w1[(size_t)n * KDIM + k] : 0.0f;
  w1h[idx] = (_Float16)v;
}

// Pack w2 [10,800] f32 -> f16 B-fragments, layout [kt][lane][16 halves],
// N zero-padded 10 -> 16.
__global__ void pack_w2_kernel(const float* __restrict__ w2,
                               _Float16* __restrict__ w2h) {
  int idx = blockIdx.x * blockDim.x + threadIdx.x;
  if (idx >= KT_TILES * 32 * 16) return;
  int h  = idx & 15;
  int t  = idx >> 4;
  int ln = t & 31;
  int kt = t >> 5;
  int n = ln & 15;
  int k = kt * 32 + kmap(ln, h);
  float v = (n < 10) ? w2[(size_t)n * NHID + k] : 0.0f;
  w2h[idx] = (_Float16)v;
}

// LIF membrane update + spike emission for one accumulator tile.
__device__ __forceinline__ void lif_update(float* __restrict__ m1row,
                                           const v8f& acc, int nt, int lane,
                                           _Float16 (*spk)[NHID]) {
  int n  = nt * 16 + (lane & 15);
  int hi = lane >> 4;
  #pragma unroll
  for (int r = 0; r < 8; ++r) {
    float mv  = LEAK * m1row[r] + (1.0f - LEAK) * acc[r];
    float thr = mv - 1.0f;                   // THRESHOLD = 1
    float o   = (thr > 0.0f) ? 1.0f : 0.0f;
    m1row[r]  = mv - o;                      // soft reset by THRESHOLD
    spk[r + (hi << 3)][n] = (_Float16)o;     // {0,1}: exact in f16
  }
}

__global__ __launch_bounds__(THREADS)
void snn_persistent_kernel(const float* __restrict__ inp,
                           const _Float16* __restrict__ w1h,
                           const _Float16* __restrict__ w2h,
                           const int* __restrict__ nsteps_p,
                           float* __restrict__ out) {
  // LDS: spikes in A-fragment layout + hidden spike matrix (row-major f16)
  __shared__ __align__(32) _Float16 spA[KT_TILES][32][16];  // 25.0 KB
  __shared__ __align__(32) _Float16 spk[BT][NHID];          // 25.0 KB

  const int tid  = threadIdx.x;
  const int wid  = tid >> 5;
  const int lane = tid & 31;
  const int row0 = blockIdx.x * BT;
  const int num_steps = nsteps_p[0];

  // Persistent membrane state in VGPRs: wave owns N-tiles {wid, wid+8, ...}.
  // Waves 0,1 own 7 tiles; waves 2..7 own 6. All indexing compile-time.
  float m1[7][8];
  #pragma unroll
  for (int i = 0; i < 7; ++i)
    #pragma unroll
    for (int r = 0; r < 8; ++r) m1[i][r] = 0.0f;
  v8f acc2 = {};  // mem2 accumulator (wave 0 only meaningful)

  const v16h* spAv = (const v16h*)&spA[0][0][0];
  const v16h* w1v  = (const v16h*)w1h;
  const v16h* w2v  = (const v16h*)w2h;

  for (int step = 0; step < num_steps; ++step) {
    // ---- Phase 1: Poisson spike gen directly into WMMA A-fragment layout
    for (int idx = tid; idx < KT_TILES * 32 * 16; idx += THREADS) {
      int h  = idx & 15;
      int ln = (idx >> 4) & 31;
      int kt = idx >> 9;
      int m  = ln & 15;
      int k  = kt * 32 + kmap(ln, h);
      float sp = 0.0f;                       // exact zero for K padding
      if (k < KDIM) {
        float x = inp[(size_t)(row0 + m) * KDIM + k];
        float r = rng01((uint32_t)(row0 + m), (uint32_t)k, (uint32_t)step);
        float t = (2.0f * r <= fabsf(x)) ? ((x >= 0.0f) ? 1.0f : -1.0f) : 0.0f;
        sp = 0.5f * (t + 1.0f);              // {0, 0.5, 1}: exact in f16
      }
      spA[kt][ln][h] = (_Float16)sp;
    }
    __syncthreads();

    // ---- Phase 2: mem1 = leak*mem1 + (1-leak)*(sp @ w1^T); spike + reset.
    // Two independent accumulator chains per pass fill the f16-WMMA
    // RAW-hazard coexec slots with real work instead of v_nop.
    #pragma unroll
    for (int p = 0; p < 3; ++p) {
      const int nt0 = wid + WAVES * (2 * p);      // <= 47
      const int nt1 = nt0 + WAVES;
      const v16h* b0p = w1v + (size_t)nt0 * KT_TILES * 32 + lane;
      const v16h* b1p = w1v + (size_t)nt1 * KT_TILES * 32 + lane;
      if (p < 2)  // prefetch next pair's B stream (L2-resident, hides latency)
        __builtin_prefetch((const void*)(b0p + (size_t)2 * WAVES * KT_TILES * 32), 0, 1);
      v8f acc0 = {}, acc1 = {};
      #pragma unroll 5
      for (int kt = 0; kt < KT_TILES; ++kt) {
        v16h a  = spAv[kt * 32 + lane];
        v16h b0 = b0p[kt * 32];               // immediate-offset b128 pairs
        v16h b1 = b1p[kt * 32];
        acc0 = WMMA_F16(a, b0, acc0);
        acc1 = WMMA_F16(a, b1, acc1);
      }
      lif_update(m1[2 * p],     acc0, nt0, lane, spk);
      lif_update(m1[2 * p + 1], acc1, nt1, lane, spk);
    }
    if (wid < 2) {  // tail tile: nt = wid + 48
      const int nt = wid + 48;
      const v16h* bp = w1v + (size_t)nt * KT_TILES * 32 + lane;
      v8f acc = {};
      #pragma unroll 5
      for (int kt = 0; kt < KT_TILES; ++kt) {
        v16h a = spAv[kt * 32 + lane];
        acc = WMMA_F16(a, bp[kt * 32], acc);
      }
      lif_update(m1[6], acc, nt, lane, spk);
    }
    __syncthreads();

    // ---- Phase 3: mem2 += out @ w2^T  (wave 0; N padded 10->16)
    if (wid == 0) {
      v8f d = {};
      int m = lane & 15;
      #pragma unroll 5
      for (int kt = 0; kt < KT_TILES; ++kt) {
        v16h a;
        #pragma unroll
        for (int h = 0; h < 16; ++h)
          a[h] = spk[m][kt * 32 + kmap(lane, h)];
        v16h b = w2v[kt * 32 + lane];
        d = WMMA_F16(a, b, d);
      }
      #pragma unroll
      for (int r = 0; r < 8; ++r) acc2[r] += d[r];
    }
    // next iteration's Phase-1 sync orders wave0's spk reads vs new writes
  }

  if (wid == 0) {
    int n  = lane & 15;
    int hi = lane >> 4;
    float scale = 1.0f / (float)num_steps;
    if (n < 10) {
      #pragma unroll
      for (int r = 0; r < 8; ++r) {
        int m = r + (hi << 3);
        out[(size_t)(row0 + m) * 10 + n] = acc2[r] * scale;
      }
    }
  }
}

extern "C" void kernel_launch(void* const* d_in, const int* in_sizes, int n_in,
                              void* d_out, int out_size, void* d_ws, size_t ws_size,
                              hipStream_t stream) {
  const float* inp    = (const float*)d_in[0];   // [8192, 784]
  const float* w1     = (const float*)d_in[1];   // [800, 784]
  const float* w2     = (const float*)d_in[2];   // [10, 800]
  const int*   nsteps = (const int*)d_in[3];     // scalar 25
  float* out = (float*)d_out;                    // [8192, 10]

  // Workspace: packed f16 weight fragments (1.28 MB + 25.6 KB)
  _Float16* w1h = (_Float16*)d_ws;
  _Float16* w2h = w1h + (size_t)NT_TILES * KT_TILES * 32 * 16;

  int n1 = NT_TILES * KT_TILES * 32 * 16;  // 640000
  pack_w1_kernel<<<(n1 + 255) / 256, 256, 0, stream>>>(w1, w1h);
  int n2 = KT_TILES * 32 * 16;             // 12800
  pack_w2_kernel<<<(n2 + 255) / 256, 256, 0, stream>>>(w2, w2h);

  snn_persistent_kernel<<<BATCH / BT, THREADS, 0, stream>>>(
      inp, w1h, w2h, nsteps, out);
}
